// EGNNModule_13048110645902
// MI455X (gfx1250) — matrix operand order repaired and checked by reading, not compiled, not run on hardware
//
#include <hip/hip_runtime.h>

// ---------------------------------------------------------------------------
// EGNN layer for MI455X (gfx1250, wave32).  Edge-MLP GEMMs dominate (~138
// GFLOP) and everything fits in L2, so the kernel is matrix-unit bound:
// we run all four MLP GEMMs through v_wmma_f32_16x16x32_bf16.
// ---------------------------------------------------------------------------

typedef __attribute__((ext_vector_type(16))) __bf16 v16bf;
typedef __attribute__((ext_vector_type(8)))  float  v8f;

#define B_     4
#define N_     2048
#define D_     256
#define K_     16
#define M_     64

#define EIN_PAD   544     // 513 -> 17 K-tiles of 32
#define EHID_PAD  1056    // 1026 -> 33 K-tiles of 32 / 66 N-tiles of 16
#define NKT1      17
#define NCT1      66
#define NKT2      33
#define NCT2      4

#define NIN_PAD   320     // 256+64, already multiple of 32
#define NH_PAD    512
#define NKT1N     10
#define NCT1N     32
#define NKT2N     16
#define NCT2N     16

__device__ __forceinline__ unsigned short f2bf_raw(float f) {
  union { float f; unsigned int u; } x; x.f = f;
  unsigned int u = x.u;
  unsigned int r = u + 0x7FFFu + ((u >> 16) & 1u);   // round-to-nearest-even
  return (unsigned short)(r >> 16);
}
__device__ __forceinline__ float silu_f(float x)    { return x / (1.0f + __expf(-x)); }
__device__ __forceinline__ float sigmoid_f(float x) { return 1.0f / (1.0f + __expf(-x)); }

// ---------------------------------------------------------------------------
// Pre-pack a row-major f32 weight [Kact x Cact] into WMMA bf16 B-fragment
// tile order: dst[((kt*numCT+ct)*32 + lane)*8 + v] = pack(bf(K=kb, col),
// bf(K=kb+1, col)), kb = (v>>2)*16 + g*8 + 2*(v&3), g = lane>>4, col = ct*16 + (lane&15).  Zero padded.
// ---------------------------------------------------------------------------
__global__ void pack_b_tiles(const float* __restrict__ src,
                             unsigned int* __restrict__ dst,
                             int Kact, int Cact, int numCT) {
  int tile = blockIdx.x;                   // kt*numCT + ct
  int kt = tile / numCT, ct = tile % numCT;
  int lane = threadIdx.x;                  // 0..31
  int g = lane >> 4, n = lane & 15;
  int c = ct * 16 + n;
  unsigned int* out = dst + ((size_t)tile * 32 + lane) * 8;
#pragma unroll
  for (int v = 0; v < 8; ++v) {
    int k0 = kt * 32 + (v >> 2) * 16 + g * 8 + 2 * (v & 3);
    float lo = (k0     < Kact && c < Cact) ? src[(size_t)k0       * Cact + c] : 0.0f;
    float hi = (k0 + 1 < Kact && c < Cact) ? src[(size_t)(k0 + 1) * Cact + c] : 0.0f;
    out[v] = (unsigned int)f2bf_raw(lo) | ((unsigned int)f2bf_raw(hi) << 16);
  }
}

// ---------------------------------------------------------------------------
// Top-K=16 nearest neighbors per node.  One wave32 per node; all N dists in
// LDS, then 16 min-extractions with wave shuffle reductions.
// ---------------------------------------------------------------------------
__global__ __launch_bounds__(32) void topk_kernel(
    const float* __restrict__ coors, const unsigned char* __restrict__ mask,
    int* __restrict__ idx_out, float* __restrict__ rd_out) {
  __shared__ float dist_lds[N_];
  __shared__ float rank_lds[N_];
  int node = blockIdx.x;                 // b*N + i
  int b = node >> 11;
  int lane = threadIdx.x;
  float cx = coors[node * 3 + 0], cy = coors[node * 3 + 1], cz = coors[node * 3 + 2];
  bool mi = mask[node] != 0;
  const float* cb = coors + (size_t)b * N_ * 3;
  const unsigned char* mb = mask + (size_t)b * N_;
  for (int j = lane; j < N_; j += 32) {
    float dx = cx - cb[j * 3 + 0];
    float dy = cy - cb[j * 3 + 1];
    float dz = cz - cb[j * 3 + 2];
    float d = dx * dx + dy * dy + dz * dz;
    dist_lds[j] = d;
    rank_lds[j] = (mi && (mb[j] != 0)) ? d : 1e5f;   // MASK_FILL
  }
  __syncthreads();
  for (int kk = 0; kk < K_; ++kk) {
    float bd = 3.4e38f; int bj = 0x7FFFFFFF;
    for (int j = lane; j < N_; j += 32) {
      float r = rank_lds[j];
      if (r < bd) { bd = r; bj = j; }
    }
#pragma unroll
    for (int off = 16; off > 0; off >>= 1) {
      float od = __shfl_xor(bd, off, 32);
      int   oj = __shfl_xor(bj, off, 32);
      if (od < bd || (od == bd && oj < bj)) { bd = od; bj = oj; }
    }
    if (lane == 0) {
      idx_out[node * K_ + kk] = bj;
      rd_out[node * K_ + kk]  = dist_lds[bj];
      rank_lds[bj] = 3.4e38f;
    }
    __syncthreads();
  }
}

// ---------------------------------------------------------------------------
// Edge MLP + gate + masked mean pool.  One 4-wave workgroup per node: the
// 16-row WMMA tile is exactly the node's K=16 edges, so pooling over tile
// rows yields m_i directly.
// ---------------------------------------------------------------------------
__global__ __launch_bounds__(128) void edge_kernel(
    const float* __restrict__ feats, const unsigned char* __restrict__ mask,
    const int* __restrict__ idx, const float* __restrict__ rd,
    const unsigned int* __restrict__ w1t, const float* __restrict__ b_e1,
    const unsigned int* __restrict__ w2t, const float* __restrict__ b_e2,
    const float* __restrict__ w_g, const float* __restrict__ b_g,
    float* __restrict__ m_i_out) {
  __shared__ __align__(16) unsigned short Xs[16 * EIN_PAD];
  __shared__ __align__(16) unsigned short Hs[16 * EHID_PAD];
  __shared__ __align__(16) float Ms[16 * M_];
  __shared__ int   nb[16];
  __shared__ float emask[16];
  __shared__ float gate[16];

  int node = blockIdx.x;                  // b*N + i
  int b = node >> 11;
  int tid = threadIdx.x;
  int wave = tid >> 5, lane = tid & 31;
  int g = lane >> 4, n = lane & 15;

  if (tid < 16) nb[tid] = idx[node * K_ + tid];
  __syncthreads();

  // Build X tile rows = K edges: [feats_i | feats_j | rel_dist | 0 pad]
  const float* fi = feats + (size_t)node * D_;
  for (int c = tid; c < D_; c += 128) {
    unsigned short v = f2bf_raw(fi[c]);
#pragma unroll
    for (int r = 0; r < 16; ++r) Xs[r * EIN_PAD + c] = v;
  }
  for (int r = wave; r < 16; r += 4) {
    const float* fj = feats + ((size_t)b * N_ + nb[r]) * D_;
    for (int c = lane; c < D_; c += 32)
      Xs[r * EIN_PAD + D_ + c] = f2bf_raw(fj[c]);
  }
  if (tid < 16) {
    int r = tid;
    Xs[r * EIN_PAD + 2 * D_] = f2bf_raw(rd[node * K_ + r]);
    for (int c = 2 * D_ + 1; c < EIN_PAD; ++c) Xs[r * EIN_PAD + c] = 0;
    bool mi = mask[node] != 0;
    bool mj = mask[(size_t)b * N_ + nb[r]] != 0;
    emask[r] = (mi && mj) ? 1.0f : 0.0f;
  }
  __syncthreads();

  // ---- GEMM1: X[16x544] @ w_e1[544x1056] -> SiLU -> Hs (bf16) ----
  for (int ct = wave; ct < NCT1; ct += 4) {
    v8f acc = {};
    for (int kt = 0; kt < NKT1; ++kt) {
      v16bf A; unsigned int* ap = (unsigned int*)&A;
      const unsigned short* xr = &Xs[n * EIN_PAD + kt * 32 + g * 8];
#pragma unroll
      for (int v = 0; v < 4; ++v) {
        ap[v]     = *(const unsigned int*)(xr + 2 * v);
        ap[v + 4] = *(const unsigned int*)(xr + 16 + 2 * v);
      }
      v16bf Bf; unsigned int* bp = (unsigned int*)&Bf;
      const unsigned int* ws = w1t + (((size_t)kt * NCT1 + ct) * 32 + lane) * 8;
#pragma unroll
      for (int v = 0; v < 8; ++v) bp[v] = ws[v];
      if (kt + 1 < NKT1)
        __builtin_prefetch(w1t + (((size_t)(kt + 1) * NCT1 + ct) * 32 + lane) * 8, 0, 1);
      acc = __builtin_amdgcn_wmma_f32_16x16x32_bf16(false, A, false, Bf,
                                                    (short)0, acc, false, false);
    }
    int col = ct * 16 + n;
    float bias = (col < 1026) ? b_e1[col] : 0.0f;
#pragma unroll
    for (int v = 0; v < 8; ++v) {
      int row = g * 8 + v;
      Hs[row * EHID_PAD + col] = f2bf_raw(silu_f(acc[v] + bias));
    }
  }
  __syncthreads();

  // ---- GEMM2: H[16x1056] @ w_e2[1056x64] -> SiLU -> Ms (f32) ----
  {
    int ct2 = wave;                                    // 64 cols = 4 tiles
    v8f acc = {};
    for (int kt = 0; kt < NKT2; ++kt) {
      v16bf A; unsigned int* ap = (unsigned int*)&A;
      const unsigned short* hr = &Hs[n * EHID_PAD + kt * 32 + g * 8];
#pragma unroll
      for (int v = 0; v < 4; ++v) {
        ap[v]     = *(const unsigned int*)(hr + 2 * v);
        ap[v + 4] = *(const unsigned int*)(hr + 16 + 2 * v);
      }
      v16bf Bf; unsigned int* bp = (unsigned int*)&Bf;
      const unsigned int* ws = w2t + (((size_t)kt * NCT2 + ct2) * 32 + lane) * 8;
#pragma unroll
      for (int v = 0; v < 8; ++v) bp[v] = ws[v];
      if (kt + 1 < NKT2)
        __builtin_prefetch(w2t + (((size_t)(kt + 1) * NCT2 + ct2) * 32 + lane) * 8, 0, 1);
      acc = __builtin_amdgcn_wmma_f32_16x16x32_bf16(false, A, false, Bf,
                                                    (short)0, acc, false, false);
    }
    int col = ct2 * 16 + n;
    float bias = b_e2[col];
#pragma unroll
    for (int v = 0; v < 8; ++v)
      Ms[(g * 8 + v) * M_ + col] = silu_f(acc[v] + bias);
  }
  __syncthreads();

  // ---- soft gate per edge row, then masked mean over rows -> m_i ----
  if (tid < 16) {
    float s = b_g[0];
    for (int c = 0; c < M_; ++c) s += Ms[tid * M_ + c] * w_g[c];
    gate[tid] = sigmoid_f(s) * emask[tid];
  }
  __syncthreads();
  if (tid < M_) {
    float denom = 0.0f;
    for (int r = 0; r < 16; ++r) denom += emask[r];
    denom = fmaxf(denom, 1.0f);
    float s = 0.0f;
    for (int r = 0; r < 16; ++r) s += Ms[r * M_ + tid] * gate[r];
    m_i_out[(size_t)node * M_ + tid] = s / denom;
  }
}

// ---------------------------------------------------------------------------
// Node MLP: silu([feats|m_i] @ w_n1) @ w_n2 + b_n2 + feats.  16 nodes per WG.
// ---------------------------------------------------------------------------
__global__ __launch_bounds__(128) void node_kernel(
    const float* __restrict__ feats, const float* __restrict__ m_i,
    const unsigned int* __restrict__ w1t, const float* __restrict__ b_n1,
    const unsigned int* __restrict__ w2t, const float* __restrict__ b_n2,
    float* __restrict__ out) {
  __shared__ __align__(16) unsigned short Xs[16 * NIN_PAD];
  __shared__ __align__(16) unsigned short Hs[16 * NH_PAD];
  int tid = threadIdx.x;
  int wave = tid >> 5, lane = tid & 31;
  int g = lane >> 4, n = lane & 15;
  size_t row0 = (size_t)blockIdx.x * 16;

  for (int r = wave; r < 16; r += 4) {
    const float* f = feats + (row0 + r) * D_;
    for (int c = lane; c < D_; c += 32) Xs[r * NIN_PAD + c] = f2bf_raw(f[c]);
    const float* mm = m_i + (row0 + r) * M_;
    for (int c = lane; c < M_; c += 32) Xs[r * NIN_PAD + D_ + c] = f2bf_raw(mm[c]);
  }
  __syncthreads();

  for (int ct = wave; ct < NCT1N; ct += 4) {
    v8f acc = {};
    for (int kt = 0; kt < NKT1N; ++kt) {
      v16bf A; unsigned int* ap = (unsigned int*)&A;
      const unsigned short* xr = &Xs[n * NIN_PAD + kt * 32 + g * 8];
#pragma unroll
      for (int v = 0; v < 4; ++v) {
        ap[v]     = *(const unsigned int*)(xr + 2 * v);
        ap[v + 4] = *(const unsigned int*)(xr + 16 + 2 * v);
      }
      v16bf Bf; unsigned int* bp = (unsigned int*)&Bf;
      const unsigned int* ws = w1t + (((size_t)kt * NCT1N + ct) * 32 + lane) * 8;
#pragma unroll
      for (int v = 0; v < 8; ++v) bp[v] = ws[v];
      acc = __builtin_amdgcn_wmma_f32_16x16x32_bf16(false, A, false, Bf,
                                                    (short)0, acc, false, false);
    }
    int col = ct * 16 + n;
    float bias = b_n1[col];
#pragma unroll
    for (int v = 0; v < 8; ++v)
      Hs[(g * 8 + v) * NH_PAD + col] = f2bf_raw(silu_f(acc[v] + bias));
  }
  __syncthreads();

  for (int ct = wave; ct < NCT2N; ct += 4) {
    v8f acc = {};
    for (int kt = 0; kt < NKT2N; ++kt) {
      v16bf A; unsigned int* ap = (unsigned int*)&A;
      const unsigned short* hr = &Hs[n * NH_PAD + kt * 32 + g * 8];
#pragma unroll
      for (int v = 0; v < 4; ++v) {
        ap[v]     = *(const unsigned int*)(hr + 2 * v);
        ap[v + 4] = *(const unsigned int*)(hr + 16 + 2 * v);
      }
      v16bf Bf; unsigned int* bp = (unsigned int*)&Bf;
      const unsigned int* ws = w2t + (((size_t)kt * NCT2N + ct) * 32 + lane) * 8;
#pragma unroll
      for (int v = 0; v < 8; ++v) bp[v] = ws[v];
      acc = __builtin_amdgcn_wmma_f32_16x16x32_bf16(false, A, false, Bf,
                                                    (short)0, acc, false, false);
    }
    int col = ct * 16 + n;
    float bias = b_n2[col];
#pragma unroll
    for (int v = 0; v < 8; ++v) {
      size_t r = row0 + g * 8 + v;
      out[r * D_ + col] = acc[v] + bias + feats[r * D_ + col];
    }
  }
}

// ---------------------------------------------------------------------------
// Passthrough: coors and mask (as float) appended after node_out.
// ---------------------------------------------------------------------------
__global__ void passthru_kernel(const float* __restrict__ coors,
                                const unsigned char* __restrict__ mask,
                                float* __restrict__ out) {
  int t = blockIdx.x * blockDim.x + threadIdx.x;
  const int NC = B_ * N_ * 3;
  float* tail = out + (size_t)B_ * N_ * D_;
  if (t < NC) tail[t] = coors[t];
  if (t < B_ * N_) tail[NC + t] = mask[t] ? 1.0f : 0.0f;
}

// ---------------------------------------------------------------------------
extern "C" void kernel_launch(void* const* d_in, const int* in_sizes, int n_in,
                              void* d_out, int out_size, void* d_ws, size_t ws_size,
                              hipStream_t stream) {
  const float*         feats = (const float*)d_in[0];
  const float*         coors = (const float*)d_in[1];
  const unsigned char* mask  = (const unsigned char*)d_in[2];
  const float* w_e1 = (const float*)d_in[3];
  const float* b_e1 = (const float*)d_in[4];
  const float* w_e2 = (const float*)d_in[5];
  const float* b_e2 = (const float*)d_in[6];
  const float* w_g  = (const float*)d_in[7];
  const float* b_g  = (const float*)d_in[8];
  const float* w_n1 = (const float*)d_in[9];
  const float* b_n1 = (const float*)d_in[10];
  const float* w_n2 = (const float*)d_in[11];
  const float* b_n2 = (const float*)d_in[12];
  float* out = (float*)d_out;

  // workspace carve-up
  char* p = (char*)d_ws;
  size_t off = 0;
  int*   idx = (int*)(p + off);   off += (size_t)B_ * N_ * K_ * 4;
  float* rd  = (float*)(p + off); off += (size_t)B_ * N_ * K_ * 4;
  float* mi  = (float*)(p + off); off += (size_t)B_ * N_ * M_ * 4;
  unsigned int* w1e = (unsigned int*)(p + off); off += (size_t)NKT1  * NCT1  * 256 * 4;
  unsigned int* w2e = (unsigned int*)(p + off); off += (size_t)NKT2  * NCT2  * 256 * 4;
  unsigned int* w1n = (unsigned int*)(p + off); off += (size_t)NKT1N * NCT1N * 256 * 4;
  unsigned int* w2n = (unsigned int*)(p + off); off += (size_t)NKT2N * NCT2N * 256 * 4;

  // 1) pack weights into WMMA fragment tiles (bf16)
  pack_b_tiles<<<NKT1  * NCT1,  32, 0, stream>>>(w_e1, w1e, 513,  1026, NCT1);
  pack_b_tiles<<<NKT2  * NCT2,  32, 0, stream>>>(w_e2, w2e, 1026, 64,   NCT2);
  pack_b_tiles<<<NKT1N * NCT1N, 32, 0, stream>>>(w_n1, w1n, 320,  512,  NCT1N);
  pack_b_tiles<<<NKT2N * NCT2N, 32, 0, stream>>>(w_n2, w2n, 512,  256,  NCT2N);

  // 2) coors/mask passthrough outputs
  passthru_kernel<<<(B_ * N_ * 3 + 255) / 256, 256, 0, stream>>>(coors, mask, out);

  // 3) kNN top-16 per node
  topk_kernel<<<B_ * N_, 32, 0, stream>>>(coors, mask, idx, rd);

  // 4) edge MLP + gate + pool  (one WG per node)
  edge_kernel<<<B_ * N_, 128, 0, stream>>>(feats, mask, idx, rd,
                                           w1e, b_e1, w2e, b_e2, w_g, b_g, mi);

  // 5) node MLP + residual  (16 nodes per WG)
  node_kernel<<<(B_ * N_) / 16, 128, 0, stream>>>(feats, mi, w1n, b_n1, w2n, b_n2, out);
}
